// AttentiveRouter_44719199486756
// MI455X (gfx1250) — matrix operand order repaired and checked by reading, not compiled, not run on hardware
//
#include <hip/hip_runtime.h>
#include <math.h>

// ---------------- types ----------------
typedef __attribute__((ext_vector_type(16))) __bf16         v16bf;
typedef __attribute__((ext_vector_type(8)))  float          v8f;
typedef __attribute__((ext_vector_type(8)))  unsigned short u16x8;
typedef __attribute__((ext_vector_type(4)))  unsigned short u16x4;
typedef __attribute__((ext_vector_type(4)))  float          f32x4;

#define D_DIM 2048
#define M_TOK 8192
#define E_NUM 64

union FragBF {
    v16bf bf;
    u16x8 h[2];
};

__device__ __forceinline__ unsigned short f2bf(float f) {
    union { float f; unsigned int u; } c; c.f = f;
    unsigned int r = c.u + 0x7FFFu + ((c.u >> 16) & 1u);   // round-to-nearest-even
    return (unsigned short)(r >> 16);
}

__device__ __forceinline__ v8f zero8() {
    v8f z = {0.f, 0.f, 0.f, 0.f, 0.f, 0.f, 0.f, 0.f};
    return z;
}

// ---------------- kernel 1: h = GELU(x @ W1 + b1), bf16 out ----------------
// grid (N/128=16, M/128=64), 256 threads (8 waves). 128x128 tile, K-tile 32.
__global__ __launch_bounds__(256)
void gemm1_gelu_kernel(const float* __restrict__ x,
                       const float* __restrict__ W1,
                       const float* __restrict__ b1,
                       unsigned short* __restrict__ hout) {
    __shared__ unsigned short As[128 * 40];   // [row][k], stride 40 (pad)
    __shared__ unsigned short Bs[128 * 40];   // [col][k], transposed W1 tile

    const int t    = threadIdx.x;
    const int lane = t & 31;
    const int w    = t >> 5;
    const int wr   = w >> 1;      // 0..3 : 32-row group
    const int wc   = w & 1;       // 0..1 : 64-col group
    const int l15  = lane & 15;
    const int kh   = lane >> 4;   // K-half selector per documented 16-bit layouts
    const int m0   = blockIdx.y * 128;
    const int n0   = blockIdx.x * 128;

    v8f acc[2][4];
#pragma unroll
    for (int mt = 0; mt < 2; ++mt)
#pragma unroll
        for (int nt = 0; nt < 4; ++nt) acc[mt][nt] = zero8();

    const int bn = t & 127;   // B-fill: column
    const int bg = t >> 7;    // B-fill: k half (0/1)

    for (int k0 = 0; k0 < D_DIM; k0 += 32) {
        // ---- A tile 128x32 fp32 -> bf16 LDS (b64 stores) ----
#pragma unroll
        for (int j = 0; j < 4; ++j) {
            int s   = t + j * 256;
            int row = s >> 3;
            int kf  = (s & 7) << 2;
            f32x4 v = *(const f32x4*)(x + (size_t)(m0 + row) * D_DIM + k0 + kf);
            u16x4 p;
            p.x = f2bf(v.x); p.y = f2bf(v.y); p.z = f2bf(v.z); p.w = f2bf(v.w);
            *(u16x4*)&As[row * 40 + kf] = p;
        }
        // ---- B tile 32x128 fp32 -> transposed bf16 LDS [n][k] (b128 stores) ----
        u16x8 pb0, pb1;
#pragma unroll
        for (int i = 0; i < 8; ++i) {
            pb0[i] = f2bf(W1[(size_t)(k0 + bg * 16 + i)     * D_DIM + n0 + bn]);
            pb1[i] = f2bf(W1[(size_t)(k0 + bg * 16 + 8 + i) * D_DIM + n0 + bn]);
        }
        *(u16x8*)&Bs[bn * 40 + bg * 16]     = pb0;
        *(u16x8*)&Bs[bn * 40 + bg * 16 + 8] = pb1;

        __syncthreads();

        // ---- fragments + 8 WMMAs ----
        FragBF af[2];
#pragma unroll
        for (int mt = 0; mt < 2; ++mt) {
            int row = wr * 32 + mt * 16 + l15;
            af[mt].h[0] = *(const u16x8*)&As[row * 40 + kh * 8];        // K = kh*8 .. +7
            af[mt].h[1] = *(const u16x8*)&As[row * 40 + 16 + kh * 8];   // K = 16+kh*8 .. +7
        }
#pragma unroll
        for (int nt = 0; nt < 4; ++nt) {
            FragBF bfr;
            int col = wc * 64 + nt * 16 + l15;
            bfr.h[0] = *(const u16x8*)&Bs[col * 40 + kh * 8];           // K = kh*8 .. +7
            bfr.h[1] = *(const u16x8*)&Bs[col * 40 + 16 + kh * 8];      // K = 16+kh*8 .. +7
#pragma unroll
            for (int mt = 0; mt < 2; ++mt) {
                acc[mt][nt] = __builtin_amdgcn_wmma_f32_16x16x32_bf16(
                    false, af[mt].bf, false, bfr.bf, (short)0, acc[mt][nt],
                    false, false);
            }
        }
        __syncthreads();
    }

    // ---- epilogue: +bias, exact GELU, store bf16 h ----
#pragma unroll
    for (int nt = 0; nt < 4; ++nt) {
        int   col  = n0 + wc * 64 + nt * 16 + l15;
        float bias = b1[col];
#pragma unroll
        for (int mt = 0; mt < 2; ++mt) {
            int rbase = m0 + wr * 32 + mt * 16 + kh * 8;
#pragma unroll
            for (int v = 0; v < 8; ++v) {
                float val = acc[mt][nt][v] + bias;
                float g   = 0.5f * val * (1.0f + erff(val * 0.70710678118654752f));
                hout[(size_t)(rbase + v) * D_DIM + col] = f2bf(g);
            }
        }
    }
}

// ---------------- kernel 2: scores -> softmax -> top2 -> outputs ----------------
// grid 64 blocks x 256 threads; each wave: 16 tokens x 64 experts.
__global__ __launch_bounds__(256)
void router_head_kernel(const unsigned short* __restrict__ hin,
                        const float* __restrict__ W2,
                        const float* __restrict__ b2,
                        const float* __restrict__ temperature,
                        float* __restrict__ mask_out,
                        float* __restrict__ attn_out,
                        float* __restrict__ idx_out,
                        float* __restrict__ usage_g) {
    __shared__ unsigned short Bs[64 * 40];      // W2 tile, [e][k] bf16
    __shared__ float sc_s[128 * 68];            // scores staging
    __shared__ float usage_s[64];

    const int t    = threadIdx.x;
    const int lane = t & 31;
    const int w    = t >> 5;
    const int l15  = lane & 15;
    const int kh   = lane >> 4;
    const int m0   = blockIdx.x * 128;

    if (t < 64) usage_s[t] = 0.0f;

    v8f acc[4];
#pragma unroll
    for (int nt = 0; nt < 4; ++nt) acc[nt] = zero8();

    const int be = t & 63;   // expert column for W2 fill
    const int bg = t >> 6;   // 0..3 k-octet

    for (int k0 = 0; k0 < D_DIM; k0 += 32) {
        u16x8 p;
#pragma unroll
        for (int i = 0; i < 8; ++i)
            p[i] = f2bf(W2[(size_t)(k0 + bg * 8 + i) * E_NUM + be]);
        *(u16x8*)&Bs[be * 40 + bg * 8] = p;
        __syncthreads();

        FragBF af;
        int m = m0 + w * 16 + l15;
        af.h[0] = *(const u16x8*)&hin[(size_t)m * D_DIM + k0 + kh * 8];
        af.h[1] = *(const u16x8*)&hin[(size_t)m * D_DIM + k0 + 16 + kh * 8];

#pragma unroll
        for (int nt = 0; nt < 4; ++nt) {
            FragBF bfr;
            int col = nt * 16 + l15;
            bfr.h[0] = *(const u16x8*)&Bs[col * 40 + kh * 8];
            bfr.h[1] = *(const u16x8*)&Bs[col * 40 + 16 + kh * 8];
            acc[nt] = __builtin_amdgcn_wmma_f32_16x16x32_bf16(
                false, af.bf, false, bfr.bf, (short)0, acc[nt], false, false);
        }
        __syncthreads();
    }

    // stage scores to LDS: token-local row, expert col
#pragma unroll
    for (int nt = 0; nt < 4; ++nt) {
        int e = nt * 16 + l15;
#pragma unroll
        for (int v = 0; v < 8; ++v) {
            int tl = w * 16 + kh * 8 + v;
            sc_s[tl * 68 + e] = acc[nt][v];
        }
    }
    __syncthreads();

    if (t < 128) {
        int   m    = m0 + t;
        float invT = 1.0f / temperature[0];
        float sv[64];
        float mx = -1e30f;
#pragma unroll
        for (int e = 0; e < 64; ++e) {
            float s = (sc_s[t * 68 + e] + b2[e]) * invT;
            sv[e] = s;
            mx = fmaxf(mx, s);
        }
        float sum = 0.f;
#pragma unroll
        for (int e = 0; e < 64; ++e) { float ex = __expf(sv[e] - mx); sv[e] = ex; sum += ex; }
        float inv = 1.0f / sum;
        int i1 = 0; float v1 = -1.f;
#pragma unroll
        for (int e = 0; e < 64; ++e) {
            float a = sv[e] * inv;
            sv[e] = a;
            attn_out[(size_t)m * 64 + e] = a;
            if (a > v1) { v1 = a; i1 = e; }     // strict > : first index on ties (jax top_k)
        }
        int i2 = 0; float v2 = -1.f;
#pragma unroll
        for (int e = 0; e < 64; ++e) {
            if (e != i1 && sv[e] > v2) { v2 = sv[e]; i2 = e; }
        }
        float sw  = v1 + v2;
        float wn1 = v1 / sw, wn2 = v2 / sw;
#pragma unroll
        for (int e = 0; e < 64; ++e) {
            float mval = (e == i1) ? wn1 : ((e == i2) ? wn2 : 0.0f);
            mask_out[(size_t)m * 64 + e] = mval;
        }
        idx_out[(size_t)m * 2]     = (float)i1;
        idx_out[(size_t)m * 2 + 1] = (float)i2;
        atomicAdd(&usage_s[i1], wn1);
        atomicAdd(&usage_s[i2], wn2);
    }
    __syncthreads();
    if (t < 64) atomicAdd(&usage_g[t], usage_s[t]);
}

// ---------------- kernel 3: losses ----------------
__global__ void init_usage_kernel(float* usage) { usage[threadIdx.x] = 0.0f; }

__global__ __launch_bounds__(64)
void loss_kernel(const float* __restrict__ usage, float* __restrict__ loss_out) {
    __shared__ float red[64];
    int   t = threadIdx.x;
    float u = usage[t];

    red[t] = u; __syncthreads();
    for (int off = 32; off > 0; off >>= 1) { if (t < off) red[t] += red[t + off]; __syncthreads(); }
    float total = red[0]; __syncthreads();
    float ideal = total * (1.0f / 64.0f);

    float dv = u - ideal;
    red[t] = dv * dv; __syncthreads();
    for (int off = 32; off > 0; off >>= 1) { if (t < off) red[t] += red[t + off]; __syncthreads(); }
    float lb = red[0] * (1.0f / 64.0f); __syncthreads();

    float cv = fmaxf(u - 2560.0f, 0.0f);   // capacity = int(1.25 * 2048)
    red[t] = cv; __syncthreads();
    for (int off = 32; off > 0; off >>= 1) { if (t < off) red[t] += red[t + off]; __syncthreads(); }
    if (t == 0) loss_out[0] = lb + red[0] * (1.0f / 64.0f);
}

// ---------------- launch ----------------
extern "C" void kernel_launch(void* const* d_in, const int* in_sizes, int n_in,
                              void* d_out, int out_size, void* d_ws, size_t ws_size,
                              hipStream_t stream) {
    const float* x           = (const float*)d_in[0];
    const float* W1          = (const float*)d_in[1];
    const float* b1          = (const float*)d_in[2];
    const float* W2          = (const float*)d_in[3];
    const float* b2          = (const float*)d_in[4];
    const float* temperature = (const float*)d_in[5];

    float* out      = (float*)d_out;
    float* mask_out = out;                                     // [8192*64]
    float* loss_out = out + (size_t)M_TOK * E_NUM;             // [1]
    float* attn_out = out + (size_t)M_TOK * E_NUM + 1;         // [8192*64]
    float* idx_out  = out + 2 * (size_t)M_TOK * E_NUM + 1;     // [8192*2]

    unsigned short* hbuf = (unsigned short*)d_ws;                           // bf16 h
    float* usage = (float*)((char*)d_ws + (size_t)M_TOK * D_DIM * 2);       // [64]

    init_usage_kernel<<<1, 64, 0, stream>>>(usage);
    gemm1_gelu_kernel<<<dim3(D_DIM / 128, M_TOK / 128), 256, 0, stream>>>(x, W1, b1, hbuf);
    router_head_kernel<<<M_TOK / 128, 256, 0, stream>>>(hbuf, W2, b2, temperature,
                                                        mask_out, attn_out, idx_out, usage);
    loss_kernel<<<1, 64, 0, stream>>>(usage, loss_out);
}
